// ComplexTransformer_51092930953584
// MI455X (gfx1250) — matrix operand order repaired and checked by reading, not compiled, not run on hardware
//
#include <hip/hip_runtime.h>
#include <math.h>

// ---------------------------------------------------------------------------
// CDNA5 (gfx1250, wave32) transformer forward pass.
// All GEMM-shaped math goes through v_wmma_f32_16x16x32_f16.
// GEMM is LDS-free: weights are pre-swizzled to f16 in B-fragment order so
// each lane's fragment is one contiguous 32B global load (L2-resident reuse);
// activations are fragment-loaded as aligned float4 runs and converted inline.
// ---------------------------------------------------------------------------

typedef __attribute__((ext_vector_type(16))) _Float16 v16h;
typedef __attribute__((ext_vector_type(8)))  _Float16 v8h;
typedef __attribute__((ext_vector_type(8)))  float    v8f;

#define D_MODEL 512
#define N_HEADS 8
#define DKH     64
#define D_FF    2048
#define VOCAB   32000
#define BATCH   8
#define SEQ     512
#define MROWS   (BATCH * SEQ)   // 4096
#define LN_EPS  1e-5f
#define NEG_INF -1.0e9f

#define FLAG_RELU 1
#define FLAG_RES  2

static __device__ __forceinline__ v8f wmma16(const v16h a, const v16h b, const v8f c) {
  // D = A(16x32 f16) * B(32x16 f16) + C(16x16 f32)
  return __builtin_amdgcn_wmma_f32_16x16x32_f16(false, a, false, b, (short)0, c,
                                                false, false);
}

static __device__ __forceinline__ v8f vzero8() {
  v8f z;
#pragma unroll
  for (int i = 0; i < 8; ++i) z[i] = 0.0f;
  return z;
}

// ---------------------------------------------------------------------------
// Embedding gather * sqrt(d) + sinusoidal positional encoding (computed inline)
// ---------------------------------------------------------------------------
__global__ __launch_bounds__(128) void k_embed(const int* __restrict__ tok,
                                               const float* __restrict__ emb,
                                               float* __restrict__ out, int seq) {
  const int row = blockIdx.x;       // b*seq + pos
  const int pos = row % seq;
  const int t   = tok[row];
  const float scale = 22.62741699796952f;       // sqrt(512)
  const int d0 = threadIdx.x * 4;
#pragma unroll
  for (int j = 0; j < 4; ++j) {
    const int d = d0 + j;
    const float divb = __expf(-(float)(d & ~1) * (9.210340371976184f / 512.0f));
    const float ang  = (float)pos * divb;
    const float pe   = (d & 1) ? __cosf(ang) : __sinf(ang);
    out[(size_t)row * D_MODEL + d] = emb[(size_t)t * D_MODEL + d] * scale + pe;
  }
}

// ---------------------------------------------------------------------------
// LayerNorm over last dim (512). One wave per row, 8 rows per 256-thread block.
// ---------------------------------------------------------------------------
__global__ __launch_bounds__(256) void k_ln(const float* __restrict__ x,
                                            float* __restrict__ y,
                                            const float* __restrict__ g,
                                            const float* __restrict__ b) {
  const int lane = threadIdx.x & 31;
  const int row  = blockIdx.x * 8 + (threadIdx.x >> 5);
  const float* xr = x + (size_t)row * D_MODEL;
  float s = 0.0f, ss = 0.0f;
#pragma unroll
  for (int i = 0; i < 16; ++i) {
    const float v = xr[lane + i * 32];
    s += v; ss += v * v;
  }
#pragma unroll
  for (int o = 16; o >= 1; o >>= 1) {
    s  += __shfl_xor(s, o, 32);
    ss += __shfl_xor(ss, o, 32);
  }
  const float mean = s * (1.0f / 512.0f);
  const float var  = ss * (1.0f / 512.0f) - mean * mean;
  const float inv  = rsqrtf(var + LN_EPS);
  float* yr = y + (size_t)row * D_MODEL;
#pragma unroll
  for (int i = 0; i < 16; ++i) {
    const int d = lane + i * 32;
    yr[d] = (xr[d] - mean) * inv * g[d] + b[d];
  }
}

// ---------------------------------------------------------------------------
// Weight swizzle: W[K,N] f32 (row-major) -> Wh f16 in B-fragment order:
//   Wh[((k>>5)*(N/16) + (n>>4))*512 + lane*16 + h]
//   lane = (n&15) | (((k>>4)&1)<<4),  h = k&15
// Each thread converts one (16-k x 1-n) column strip -> one contiguous v16h.
// threads total = (K/16)*N
// ---------------------------------------------------------------------------
__global__ __launch_bounds__(256) void k_swz(const float* __restrict__ W,
                                             _Float16* __restrict__ Wh,
                                             int N, int K) {
  const size_t t   = (size_t)blockIdx.x * 256 + threadIdx.x;
  const int    n   = (int)(t % N);
  const int    k0  = (int)(t / N) * 16;
  const int    lane = (n & 15) | (((k0 >> 4) & 1) << 4);
  const size_t dst = (((size_t)(k0 >> 5) * (size_t)(N >> 4) + (size_t)(n >> 4)) * 32
                      + (size_t)lane) * 16;
  v16h v;
#pragma unroll
  for (int j = 0; j < 16; ++j) v[j] = (_Float16)W[(size_t)(k0 + j) * N + n];
  *(v16h*)&Wh[dst] = v;
}

// ---------------------------------------------------------------------------
// LDS-free WMMA GEMM: C[M,N] = act(A[M,K] @ W[K,N] + bias[N]) (+ R[M,N])
// Block = 256 threads (8 waves); tile 128x128; each wave owns 16x128.
// Per 32-k-step, per wave: one A fragment (4x float4 loads + cvt) reused by
// 8 WMMAs against 8 pre-swizzled f16 B fragments (contiguous 32B each).
//   A frag (16x32):  lane = m&15 (hi = k-half select), run [k0+hi*16, +16)
//   C frag (16x16):  lane = n | ((m>>3)<<4), vgpr = m & 7
// ---------------------------------------------------------------------------
__global__ __launch_bounds__(256) void k_gemm(const float* __restrict__ A,
                                              const _Float16* __restrict__ Wh,
                                              const float* __restrict__ bias,
                                              const float* __restrict__ R,
                                              float* __restrict__ C,
                                              int N, int K, int flags) {
  const int lane = threadIdx.x & 31;
  const int wave = threadIdx.x >> 5;
  const int nl = lane & 15, hi = lane >> 4;
  const int m0 = blockIdx.y * 128 + wave * 16;
  const int n0 = blockIdx.x * 128;
  const int ntiles = N >> 4;

  v8f acc[8];
#pragma unroll
  for (int s = 0; s < 8; ++s) acc[s] = vzero8();

  const float* ar = A + (size_t)(m0 + nl) * K + (hi << 4);
  for (int k0 = 0; k0 < K; k0 += 32, ar += 32) {
    const float4 q0 = *(const float4*)(ar);
    const float4 q1 = *(const float4*)(ar + 4);
    const float4 q2 = *(const float4*)(ar + 8);
    const float4 q3 = *(const float4*)(ar + 12);
    v16h a;
    a[0]  = (_Float16)q0.x; a[1]  = (_Float16)q0.y;
    a[2]  = (_Float16)q0.z; a[3]  = (_Float16)q0.w;
    a[4]  = (_Float16)q1.x; a[5]  = (_Float16)q1.y;
    a[6]  = (_Float16)q1.z; a[7]  = (_Float16)q1.w;
    a[8]  = (_Float16)q2.x; a[9]  = (_Float16)q2.y;
    a[10] = (_Float16)q2.z; a[11] = (_Float16)q2.w;
    a[12] = (_Float16)q3.x; a[13] = (_Float16)q3.y;
    a[14] = (_Float16)q3.z; a[15] = (_Float16)q3.w;

    const _Float16* wp = Wh + ((size_t)(k0 >> 5) * (size_t)ntiles + (size_t)(n0 >> 4)) * 512
                            + (size_t)lane * 16;
#pragma unroll
    for (int s = 0; s < 8; ++s) {
      const v16h bf = *(const v16h*)(wp + (size_t)s * 512);
      acc[s] = wmma16(a, bf, acc[s]);
    }
  }

#pragma unroll
  for (int s = 0; s < 8; ++s) {
    const int n = n0 + s * 16 + nl;
    const float bv = bias ? bias[n] : 0.0f;
#pragma unroll
    for (int r = 0; r < 8; ++r) {
      const int m = m0 + (hi << 3) + r;
      float v = acc[s][r] + bv;
      if (flags & FLAG_RELU) v = fmaxf(v, 0.0f);
      if (flags & FLAG_RES)  v += R[(size_t)m * N + n];
      C[(size_t)m * N + n] = v;
    }
  }
}

// ---------------------------------------------------------------------------
// Fused attention: one wave per (batch, head, 16-query tile).
// scores = (Q Kt)/8 [+causal mask] -> LDS f32; softmax; probs -> LDS f16 in
// row-major (fragment-friendly, aligned v8h reads); ctx = probs @ V via WMMA.
// Q/K/V/O are [B*S, 512] with head h in columns h*64..h*64+63.
// ---------------------------------------------------------------------------
__global__ __launch_bounds__(32) void k_attn(const float* __restrict__ Q,
                                             const float* __restrict__ Km,
                                             const float* __restrict__ Vm,
                                             float* __restrict__ O,
                                             int Skv, int causal) {
  __shared__ float sc[16][512];
  __shared__ alignas(32) _Float16 pr[16][512];

  const int lane = threadIdx.x;
  const int q0 = blockIdx.x * 16;
  const int h  = blockIdx.y;
  const int b  = blockIdx.z;
  const int nl = lane & 15, hi = lane >> 4;

  // --- Q fragments (16 x 64 = two 16x32 k-steps) ---
  const float* qrow = Q + ((size_t)(b * SEQ + q0 + nl)) * D_MODEL + h * DKH;
  v16h qa[2];
#pragma unroll
  for (int kf = 0; kf < 2; ++kf) {
#pragma unroll
    for (int hh = 0; hh < 16; ++hh) {
      const int k = kf * 32 + hh + (hi << 4);
      qa[kf][hh] = (_Float16)qrow[k];
    }
  }

  // --- scores ---
  for (int n0 = 0; n0 < Skv; n0 += 16) {
    v8f acc = vzero8();
    const float* kr = Km + ((size_t)(b * Skv + n0 + nl)) * D_MODEL + h * DKH + (hi << 4);
#pragma unroll
    for (int kf = 0; kf < 2; ++kf) {
      v16h bf;
#pragma unroll
      for (int hh = 0; hh < 16; ++hh) bf[hh] = (_Float16)kr[kf * 32 + hh];
      acc = wmma16(qa[kf], bf, acc);
    }
#pragma unroll
    for (int r = 0; r < 8; ++r) {
      const int m = (hi << 3) + r;
      float v = acc[r] * 0.125f;                       // 1/sqrt(64)
      if (causal && (n0 + nl) > (q0 + m)) v = NEG_INF;
      sc[m][n0 + nl] = v;
    }
  }
  __syncthreads();   // single-wave WG: compiler/LDS fence

  // --- softmax per row (32 lanes cooperate) ---
  for (int m = 0; m < 16; ++m) {
    float mx = -3.4e38f;
    for (int j = lane; j < Skv; j += 32) mx = fmaxf(mx, sc[m][j]);
#pragma unroll
    for (int o = 16; o >= 1; o >>= 1) mx = fmaxf(mx, __shfl_xor(mx, o, 32));
    float sum = 0.0f;
    for (int j = lane; j < Skv; j += 32) {
      const float e = __expf(sc[m][j] - mx);
      sum += e;
      sc[m][j] = e;
    }
#pragma unroll
    for (int o = 16; o >= 1; o >>= 1) sum += __shfl_xor(sum, o, 32);
    const float inv = 1.0f / sum;
    for (int j = lane; j < Skv; j += 32) pr[m][j] = (_Float16)(sc[m][j] * inv);
  }
  __syncthreads();

  // --- ctx = probs(16 x Skv) @ V(Skv x 64) ---
  v8f c[4];
#pragma unroll
  for (int s = 0; s < 4; ++s) c[s] = vzero8();

  for (int k0 = 0; k0 < Skv; k0 += 32) {
    const v8h p0 = *(const v8h*)&pr[nl][k0 + (hi << 4)];
    const v8h p1 = *(const v8h*)&pr[nl][k0 + 8 + (hi << 4)];
    v16h a;
#pragma unroll
    for (int i = 0; i < 8; ++i) { a[i] = p0[i]; a[8 + i] = p1[i]; }
#pragma unroll
    for (int s = 0; s < 4; ++s) {
      const float* vc = Vm + ((size_t)(b * Skv + k0 + (hi << 4))) * D_MODEL
                           + h * DKH + s * 16 + nl;
      v16h bf;
#pragma unroll
      for (int hh = 0; hh < 16; ++hh) bf[hh] = (_Float16)vc[(size_t)hh * D_MODEL];
      c[s] = wmma16(a, bf, c[s]);
    }
  }

#pragma unroll
  for (int s = 0; s < 4; ++s) {
#pragma unroll
    for (int r = 0; r < 8; ++r) {
      O[((size_t)(b * SEQ + q0 + (hi << 3) + r)) * D_MODEL + h * DKH + s * 16 + nl] =
          c[s][r];
    }
  }
}

// ---------------------------------------------------------------------------
// Host orchestration
// ---------------------------------------------------------------------------
extern "C" void kernel_launch(void* const* d_in, const int* in_sizes, int n_in,
                              void* d_out, int out_size, void* d_ws, size_t ws_size,
                              hipStream_t stream) {
  (void)in_sizes; (void)n_in; (void)out_size; (void)ws_size;

  const int* src = (const int*)d_in[0];
  const int* tgt = (const int*)d_in[1];

  // params flattened as a jax pytree: top-level dict order {src,tgt,params},
  // nested dicts in sorted-key order.
  int i = 2;
  const float* dec_b1    = (const float*)d_in[i++];  // (6,2048)
  const float* dec_b2    = (const float*)d_in[i++];  // (6,512)
  const float* dec_bk    = (const float*)d_in[i++];
  const float* dec_bo    = (const float*)d_in[i++];
  const float* dec_bq    = (const float*)d_in[i++];
  const float* dec_bv    = (const float*)d_in[i++];
  const float* dec_cbk   = (const float*)d_in[i++];
  const float* dec_cbo   = (const float*)d_in[i++];
  const float* dec_cbq   = (const float*)d_in[i++];
  const float* dec_cbv   = (const float*)d_in[i++];
  const float* dec_cwk   = (const float*)d_in[i++];  // (6,512,512)
  const float* dec_cwo   = (const float*)d_in[i++];
  const float* dec_cwq   = (const float*)d_in[i++];
  const float* dec_cwv   = (const float*)d_in[i++];
  const float* dec_ln1_b = (const float*)d_in[i++];
  const float* dec_ln1_g = (const float*)d_in[i++];
  const float* dec_ln2_b = (const float*)d_in[i++];
  const float* dec_ln2_g = (const float*)d_in[i++];
  const float* dec_ln3_b = (const float*)d_in[i++];
  const float* dec_ln3_g = (const float*)d_in[i++];
  const float* dec_w1    = (const float*)d_in[i++];  // (6,512,2048)
  const float* dec_w2    = (const float*)d_in[i++];  // (6,2048,512)
  const float* dec_wk    = (const float*)d_in[i++];
  const float* dec_wo    = (const float*)d_in[i++];
  const float* dec_wq    = (const float*)d_in[i++];
  const float* dec_wv    = (const float*)d_in[i++];
  const float* dec_norm_b = (const float*)d_in[i++];
  const float* dec_norm_g = (const float*)d_in[i++];
  const float* enc_b1    = (const float*)d_in[i++];
  const float* enc_b2    = (const float*)d_in[i++];
  const float* enc_bk    = (const float*)d_in[i++];
  const float* enc_bo    = (const float*)d_in[i++];
  const float* enc_bq    = (const float*)d_in[i++];
  const float* enc_bv    = (const float*)d_in[i++];
  const float* enc_ln1_b = (const float*)d_in[i++];
  const float* enc_ln1_g = (const float*)d_in[i++];
  const float* enc_ln2_b = (const float*)d_in[i++];
  const float* enc_ln2_g = (const float*)d_in[i++];
  const float* enc_w1    = (const float*)d_in[i++];
  const float* enc_w2    = (const float*)d_in[i++];
  const float* enc_wk    = (const float*)d_in[i++];
  const float* enc_wo    = (const float*)d_in[i++];
  const float* enc_wq    = (const float*)d_in[i++];
  const float* enc_wv    = (const float*)d_in[i++];
  const float* enc_norm_b = (const float*)d_in[i++];
  const float* enc_norm_g = (const float*)d_in[i++];
  const float* out_b     = (const float*)d_in[i++];  // (32000,)
  const float* out_w     = (const float*)d_in[i++];  // (512,32000)
  const float* src_emb   = (const float*)d_in[i++];  // (32000,512)
  const float* tgt_emb   = (const float*)d_in[i++];

  // workspace layout
  const size_t MD = (size_t)MROWS * D_MODEL;   // 2M floats
  float* ws  = (float*)d_ws;
  float* X   = ws;                             // encoder stream
  float* Y   = X + MD;                         // decoder stream
  float* Hh  = Y + MD;                         // LN output
  float* Qb  = Hh + MD;
  float* Kb  = Qb + MD;
  float* Vb  = Kb + MD;
  float* Cb  = Vb + MD;                        // attention ctx
  float* MID = Cb + MD;                        // FFN hidden, 4096x2048
  float* ENC = MID + (size_t)MROWS * D_FF;     // encoder output (post-LN)
  _Float16* WH = (_Float16*)(ENC + MD);        // f16 swizzled weights (<=33MB)

  auto gemm = [&](const float* A, const float* W, const float* bias,
                  const float* R, float* Cc, int N, int K, int flags) {
    const unsigned swz_blocks = (unsigned)(((size_t)K / 16 * (size_t)N) / 256);
    k_swz<<<swz_blocks, 256, 0, stream>>>(W, WH, N, K);
    dim3 g(N / 128, MROWS / 128);
    k_gemm<<<g, 256, 0, stream>>>(A, WH, bias, R, Cc, N, K, flags);
  };
  auto ln = [&](const float* x, float* y, const float* g_, const float* b_) {
    k_ln<<<MROWS / 8, 256, 0, stream>>>(x, y, g_, b_);
  };
  auto attn = [&](const float* Qp, const float* Kp, const float* Vp, float* Op,
                  int causal) {
    dim3 g(SEQ / 16, N_HEADS, BATCH);
    k_attn<<<g, 32, 0, stream>>>(Qp, Kp, Vp, Op, SEQ, causal);
  };

  const size_t WDD = (size_t)D_MODEL * D_MODEL;   // per-layer 512x512
  const size_t WDF = (size_t)D_MODEL * D_FF;      // per-layer 512x2048 / 2048x512

  // ---------------- encoder ----------------
  k_embed<<<MROWS, 128, 0, stream>>>(src, src_emb, X, SEQ);
  for (int l = 0; l < 6; ++l) {
    const size_t w = l * WDD, v = (size_t)l * D_MODEL;
    const size_t wf = l * WDF, vf = (size_t)l * D_FF;
    ln(X, Hh, enc_ln1_g + v, enc_ln1_b + v);
    gemm(Hh, enc_wq + w, enc_bq + v, nullptr, Qb, D_MODEL, D_MODEL, 0);
    gemm(Hh, enc_wk + w, enc_bk + v, nullptr, Kb, D_MODEL, D_MODEL, 0);
    gemm(Hh, enc_wv + w, enc_bv + v, nullptr, Vb, D_MODEL, D_MODEL, 0);
    attn(Qb, Kb, Vb, Cb, 0);
    gemm(Cb, enc_wo + w, enc_bo + v, X, X, D_MODEL, D_MODEL, FLAG_RES);
    ln(X, Hh, enc_ln2_g + v, enc_ln2_b + v);
    gemm(Hh, enc_w1 + wf, enc_b1 + vf, nullptr, MID, D_FF, D_MODEL, FLAG_RELU);
    gemm(MID, enc_w2 + wf, enc_b2 + v, X, X, D_MODEL, D_FF, FLAG_RES);
  }
  ln(X, ENC, enc_norm_g, enc_norm_b);

  // ---------------- decoder ----------------
  k_embed<<<MROWS, 128, 0, stream>>>(tgt, tgt_emb, Y, SEQ);
  for (int l = 0; l < 6; ++l) {
    const size_t w = l * WDD, v = (size_t)l * D_MODEL;
    const size_t wf = l * WDF, vf = (size_t)l * D_FF;
    // masked self-attention
    ln(Y, Hh, dec_ln1_g + v, dec_ln1_b + v);
    gemm(Hh, dec_wq + w, dec_bq + v, nullptr, Qb, D_MODEL, D_MODEL, 0);
    gemm(Hh, dec_wk + w, dec_bk + v, nullptr, Kb, D_MODEL, D_MODEL, 0);
    gemm(Hh, dec_wv + w, dec_bv + v, nullptr, Vb, D_MODEL, D_MODEL, 0);
    attn(Qb, Kb, Vb, Cb, 1);
    gemm(Cb, dec_wo + w, dec_bo + v, Y, Y, D_MODEL, D_MODEL, FLAG_RES);
    // cross-attention (K/V from encoder output)
    ln(Y, Hh, dec_ln2_g + v, dec_ln2_b + v);
    gemm(Hh,  dec_cwq + w, dec_cbq + v, nullptr, Qb, D_MODEL, D_MODEL, 0);
    gemm(ENC, dec_cwk + w, dec_cbk + v, nullptr, Kb, D_MODEL, D_MODEL, 0);
    gemm(ENC, dec_cwv + w, dec_cbv + v, nullptr, Vb, D_MODEL, D_MODEL, 0);
    attn(Qb, Kb, Vb, Cb, 0);
    gemm(Cb, dec_cwo + w, dec_cbo + v, Y, Y, D_MODEL, D_MODEL, FLAG_RES);
    // FFN
    ln(Y, Hh, dec_ln3_g + v, dec_ln3_b + v);
    gemm(Hh, dec_w1 + wf, dec_b1 + vf, nullptr, MID, D_FF, D_MODEL, FLAG_RELU);
    gemm(MID, dec_w2 + wf, dec_b2 + v, Y, Y, D_MODEL, D_FF, FLAG_RES);
  }
  ln(Y, Hh, dec_norm_g, dec_norm_b);

  // ---------------- logits ----------------
  gemm(Hh, out_w, out_b, nullptr, (float*)d_out, VOCAB, D_MODEL, 0);
}
